// MultiHeadAttention_67516885893654
// MI455X (gfx1250) — matrix operand order repaired and checked by reading, not compiled, not run on hardware
//
#include <hip/hip_runtime.h>
#include <cstdint>
#include <cstddef>

typedef __bf16 bf16_t;
typedef __attribute__((ext_vector_type(8)))  bf16_t v8bf;
typedef __attribute__((ext_vector_type(16))) bf16_t v16bf;
typedef __attribute__((ext_vector_type(8)))  float  v8f;
typedef __attribute__((ext_vector_type(4)))  uint32_t u32x4;
typedef __attribute__((ext_vector_type(8)))  int      i32x8;
typedef __attribute__((ext_vector_type(4)))  int      i32x4;

#define BB 2
#define SS 4096
#define DD 512
#define HH 8
#define HD 64

__device__ __forceinline__ bf16_t f2bf(float f) {
  uint32_t u = __builtin_bit_cast(uint32_t, f);
  uint32_t r = u + 0x7FFFu + ((u >> 16) & 1u);   // round-to-nearest-even
  uint16_t h = (uint16_t)(r >> 16);
  return __builtin_bit_cast(bf16_t, h);
}

struct bfpair { v8bf a, b; };

// A-fragment (16x32, MxK): lane holds row m=lane&15; elements 0-7 at k = koff+hi*8,
// elements 8-15 at k = koff+16+hi*8  ->  two 16B loads, 32B apart.
__device__ __forceinline__ v16bf frag_a(const bf16_t* p) {
  bfpair s{ *(const v8bf*)p, *(const v8bf*)(p + 16) };
  return __builtin_bit_cast(v16bf, s);
}
// B-fragment (32x16, KxN): lane holds col n=lane&15; elements 0-7 at k = koff+hi*8,
// elements 8-15 at k = koff+16+hi*8  ->  two 16B loads, 32B apart.
__device__ __forceinline__ v16bf frag_b(const bf16_t* p) {
  bfpair s{ *(const v8bf*)p, *(const v8bf*)(p + 16) };
  return __builtin_bit_cast(v16bf, s);
}

__device__ __forceinline__ v8f wmma_bf(v16bf a, v16bf b, v8f c) {
  return __builtin_amdgcn_wmma_f32_16x16x32_bf16(false, a, false, b, (short)0, c,
                                                 false, false);
}

// reductions across the 16 lanes of a half-wave (rows live in lane halves)
__device__ __forceinline__ float red16_max(float v, int lane) {
#pragma unroll
  for (int m = 1; m < 16; m <<= 1) {
    float o = __builtin_bit_cast(
        float, __builtin_amdgcn_ds_bpermute(((lane ^ m) << 2),
                                            __builtin_bit_cast(int, v)));
    v = fmaxf(v, o);
  }
  return v;
}
__device__ __forceinline__ float red16_sum(float v, int lane) {
#pragma unroll
  for (int m = 1; m < 16; m <<= 1) {
    float o = __builtin_bit_cast(
        float, __builtin_amdgcn_ds_bpermute(((lane ^ m) << 2),
                                            __builtin_bit_cast(int, v)));
    v += o;
  }
  return v;
}

// ---- Tensor Data Mover: DMA a 2D bf16 tile (rows x cols) into LDS ----------
// D# per ISA 8.3/8.4: group0 = {flags, lds_addr, global_addr, type=2},
// group1 = {data_size=2B, tensor dims, tile dims, dim0 stride}. 2D tensor ->
// trailing descriptor groups zeroed (tile_dim2/3 = 0 => unused).
// amdgpu-toolchain (clang-23) builtin: 6 args, last is cache policy.
__device__ __forceinline__ void tdm_load_2d_bf16(uint32_t lds_off,
                                                 const bf16_t* gptr,
                                                 uint32_t cols, uint32_t rows,
                                                 uint32_t row_stride_elems) {
  uint64_t ga = (uint64_t)(uintptr_t)gptr;
  u32x4 g0 = { 1u,                                   // count=1 (valid, user mode)
               lds_off,                              // lds_addr (bytes)
               (uint32_t)ga,                         // global_addr[31:0]
               (uint32_t)((ga >> 32) & 0x01FFFFFFu)  // global_addr[56:32]
                   | (2u << 30) };                   // type=2 ("image")
  i32x8 g1 = { (int)(1u << 16),                      // wg_mask=0, data_size=1 (2B)
               (int)((cols & 0xFFFFu) << 16),        // tensor_dim0[15:0]
               (int)(((cols >> 16) & 0xFFFFu) | ((rows & 0xFFFFu) << 16)),
               (int)(((rows >> 16) & 0xFFFFu) | ((cols & 0xFFFFu) << 16)),
               (int)(rows & 0xFFFFu),                // tile_dim1, tile_dim2=0
               (int)row_stride_elems,                // tensor_dim0_stride[31:0]
               0, 0 };                               // stride0 hi / stride1 = 0
  i32x4 gz4 = { 0, 0, 0, 0 };
  i32x8 gz8 = { 0, 0, 0, 0, 0, 0, 0, 0 };
  __builtin_amdgcn_tensor_load_to_lds(g0, g1, gz4, gz4, gz8, 0);
}

__global__ void cvt_f32_bf16(const float* __restrict__ src,
                             bf16_t* __restrict__ dst, int n) {
  int i = blockIdx.x * blockDim.x + threadIdx.x;
  if (i < n) dst[i] = f2bf(src[i]);
}

// One wave computes one 16x16 tile of one of Q/K/V = x @ W^T + b.
__global__ void __launch_bounds__(128)
qkv_proj(const bf16_t* __restrict__ xbf, const bf16_t* __restrict__ wbf,
         const float* __restrict__ bq, const float* __restrict__ bk,
         const float* __restrict__ bv, bf16_t* __restrict__ qo,
         bf16_t* __restrict__ ko, bf16_t* __restrict__ vto) {
  const int lane = threadIdx.x & 31;
  const int gw   = blockIdx.x * (blockDim.x >> 5) + (threadIdx.x >> 5);
  const int j    = gw & 31;          // output-channel chunk (16 of 512)
  const int it   = (gw >> 5) & 511;  // row tile over flattened B*S
  const int mat  = gw >> 14;         // 0=Q 1=K 2=V
  const int hi   = (lane >> 4) & 1;
  const int nn   = lane & 15;

  const bf16_t* A  = xbf + it * 16 * DD + nn * DD + hi * 8;
  const bf16_t* Bp = wbf + mat * DD * DD + (j * 16 + nn) * DD + hi * 8;

  v8f c = {};
#pragma unroll 4
  for (int kb = 0; kb < DD; kb += 32) {
    c = wmma_bf(frag_a(A + kb), frag_b(Bp + kb), c);
  }

  const float* bias = (mat == 0) ? bq : (mat == 1) ? bk : bv;
  const float bval  = bias[j * 16 + nn];
#pragma unroll
  for (int r = 0; r < 8; ++r) c[r] += bval;

  const int bidx = it >> 8;          // batch
  const int s0   = (it & 255) * 16;  // sequence offset
  const int h    = j >> 2;           // head
  const int d0   = (j & 3) * 16;     // offset within head

  if (mat < 2) {  // Q,K stored [B,H,S,HD] row-major (bf16)
    bf16_t* dst =
        (mat == 0 ? qo : ko) + ((bidx * HH + h) * SS + s0) * HD + d0 + nn;
#pragma unroll
    for (int r = 0; r < 8; ++r) dst[(r + hi * 8) * HD] = f2bf(c[r]);
  } else {        // V stored transposed [B,H,HD,S] (bf16) for P@V B-fragments
    bf16_t* dst = vto + ((bidx * HH + h) * HD + d0 + nn) * SS + s0;
#pragma unroll
    for (int r = 0; r < 8; ++r) dst[r + hi * 8] = f2bf(c[r]);
  }
}

// Block = 4 waves sharing one (b,h): wave w handles q-tile qt4*4+w.
// K/V tiles (32 keys) are DMA'd into double-buffered LDS by the TDM (wave 0),
// fenced with s_wait_tensorcnt + workgroup barriers, and consumed by all
// 4 waves (4x reuse of every K/V byte fetched).
__global__ void __launch_bounds__(128)
attn_fwd(const bf16_t* __restrict__ qbf, const bf16_t* __restrict__ kbf,
         const bf16_t* __restrict__ vtbf, float* __restrict__ out) {
  __shared__ bf16_t kT[2][32 * HD];    // [kpos_local][d]      8 KB
  __shared__ bf16_t vT[2][HD * 32];    // [d][kpos_local]      8 KB
  __shared__ bf16_t ldsP[4][16 * 32];  // per-wave P tiles     4 KB

  const int lane = threadIdx.x & 31;
  const int widx = threadIdx.x >> 5;
  const int blk  = blockIdx.x;       // B*H*(S/64) = 1024
  const int qt   = (blk & 63) * 4 + widx;
  const int h    = (blk >> 6) & 7;
  const int b    = blk >> 9;
  const int hi   = (lane >> 4) & 1;
  const int nn   = lane & 15;
  bf16_t* P = ldsP[widx];

  const bf16_t* Qb = qbf + ((size_t)(b * HH + h) * SS + qt * 16) * HD;
  const bf16_t* Kb = kbf + (size_t)(b * HH + h) * SS * HD;
  const bf16_t* Vb = vtbf + (size_t)(b * HH + h) * HD * SS;

  const uint32_t kOff[2] = { (uint32_t)(uintptr_t)&kT[0][0],
                             (uint32_t)(uintptr_t)&kT[1][0] };
  const uint32_t vOff[2] = { (uint32_t)(uintptr_t)&vT[0][0],
                             (uint32_t)(uintptr_t)&vT[1][0] };

  const v16bf aq0 = frag_a(Qb + nn * HD + hi * 8);       // d = 0..31
  const v16bf aq1 = frag_a(Qb + nn * HD + hi * 8 + 32);  // d = 32..63

  float M[8], L[8];
  v8f o0 = {}, o1 = {}, o2 = {}, o3 = {};
#pragma unroll
  for (int r = 0; r < 8; ++r) { M[r] = -__builtin_inff(); L[r] = 0.f; }

  // prologue: DMA first K/V tile into buffer 0
  if (widx == 0) {
    tdm_load_2d_bf16(kOff[0], Kb, HD, 32, HD);       // 32 keys x 64 d
    tdm_load_2d_bf16(vOff[0], Vb, 32, HD, SS);       // 64 d x 32 keys
  }

  int cur = 0;
  for (int kb = 0; kb < SS; kb += 32) {
    __syncthreads();  // all waves done reading buffer cur^1 (prev iteration)
    if (widx == 0) {
      if (kb + 32 < SS) {  // prefetch next tile into the other buffer
        tdm_load_2d_bf16(kOff[cur ^ 1], Kb + (kb + 32) * HD, HD, 32, HD);
        tdm_load_2d_bf16(vOff[cur ^ 1], Vb + (kb + 32), 32, HD, SS);
        __builtin_amdgcn_s_wait_tensorcnt(2);  // current pair complete
      } else {
        __builtin_amdgcn_s_wait_tensorcnt(0);
      }
    }
    __syncthreads();  // buffer cur published to all waves
    const bf16_t* Kt = kT[cur];
    const bf16_t* Vt = vT[cur];

    // ---- scores: S = Q @ K^T for 32 key positions (two 16-col tiles) ----
    const bf16_t* Kc0 = Kt + nn * HD + hi * 8;
    const bf16_t* Kc1 = Kt + (16 + nn) * HD + hi * 8;
    v8f s0 = {}, s1 = {};
    s0 = wmma_bf(aq0, frag_b(Kc0), s0);
    s0 = wmma_bf(aq1, frag_b(Kc0 + 32), s0);
    s1 = wmma_bf(aq0, frag_b(Kc1), s1);
    s1 = wmma_bf(aq1, frag_b(Kc1 + 32), s1);

    // ---- online softmax (rows split across lane halves) ----
    float alpha[8], p0[8], p1[8];
#pragma unroll
    for (int r = 0; r < 8; ++r) {
      float a0 = s0[r] * 0.125f;  // 1/sqrt(64)
      float a1 = s1[r] * 0.125f;
      float mx = red16_max(fmaxf(a0, a1), lane);
      float nM = fmaxf(M[r], mx);
      float al = __expf(M[r] - nM);
      float e0 = __expf(a0 - nM);
      float e1 = __expf(a1 - nM);
      float rs = red16_sum(e0 + e1, lane);
      L[r] = L[r] * al + rs;
      M[r] = nM;
      alpha[r] = al; p0[r] = e0; p1[r] = e1;
    }
#pragma unroll
    for (int r = 0; r < 8; ++r) {
      o0[r] *= alpha[r]; o1[r] *= alpha[r];
      o2[r] *= alpha[r]; o3[r] *= alpha[r];
    }

    // ---- P (16x32) C-layout -> A-layout via per-wave LDS tile ----
#pragma unroll
    for (int r = 0; r < 8; ++r) {
      int m = r + hi * 8;
      P[m * 32 + nn]      = f2bf(p0[r]);
      P[m * 32 + 16 + nn] = f2bf(p1[r]);
    }
    asm volatile("s_wait_dscnt 0" ::: "memory");
    v16bf pa = frag_a(P + nn * 32 + hi * 8);

    // ---- O += P @ V  (V tile in LDS, d-major: contiguous B-fragments) ----
    const bf16_t* Vc = Vt + hi * 8;
    o0 = wmma_bf(pa, frag_b(Vc + (0 * 16 + nn) * 32), o0);
    o1 = wmma_bf(pa, frag_b(Vc + (1 * 16 + nn) * 32), o1);
    o2 = wmma_bf(pa, frag_b(Vc + (2 * 16 + nn) * 32), o2);
    o3 = wmma_bf(pa, frag_b(Vc + (3 * 16 + nn) * 32), o3);

    cur ^= 1;
  }

  // ---- normalize and store out[b, s, h*64 + d] (fp32) ----
  float* ob = out + ((size_t)(b * SS + qt * 16)) * DD + h * HD + nn;
#pragma unroll
  for (int r = 0; r < 8; ++r) {
    int m = r + hi * 8;
    float inv = 1.0f / L[r];
    float* rowp = ob + (size_t)m * DD;
    rowp[0]  = o0[r] * inv;
    rowp[16] = o1[r] * inv;
    rowp[32] = o2[r] * inv;
    rowp[48] = o3[r] * inv;
  }
}

extern "C" void kernel_launch(void* const* d_in, const int* in_sizes, int n_in,
                              void* d_out, int out_size, void* d_ws,
                              size_t ws_size, hipStream_t stream) {
  (void)in_sizes; (void)n_in; (void)out_size; (void)ws_size;
  const float* x  = (const float*)d_in[0];
  const float* Wq = (const float*)d_in[1];
  const float* bq = (const float*)d_in[2];
  const float* Wk = (const float*)d_in[3];
  const float* bk = (const float*)d_in[4];
  const float* Wv = (const float*)d_in[5];
  const float* bv = (const float*)d_in[6];
  float* out = (float*)d_out;

  const int N_X = BB * SS * DD;  // 4,194,304 elements
  const int N_W = DD * DD;       //   262,144 elements
  bf16_t* xbf  = (bf16_t*)d_ws;
  bf16_t* wbf  = xbf + N_X;          // Wq|Wk|Wv bf16
  bf16_t* qbf  = wbf + 3 * N_W;      // [B,H,S,HD]
  bf16_t* kbf  = qbf + N_X;          // [B,H,S,HD]
  bf16_t* vtbf = kbf + N_X;          // [B,H,HD,S]
  // total workspace ~33.5 MB

  cvt_f32_bf16<<<(N_X + 255) / 256, 256, 0, stream>>>(x, xbf, N_X);
  cvt_f32_bf16<<<(N_W + 255) / 256, 256, 0, stream>>>(Wq, wbf, N_W);
  cvt_f32_bf16<<<(N_W + 255) / 256, 256, 0, stream>>>(Wk, wbf + N_W, N_W);
  cvt_f32_bf16<<<(N_W + 255) / 256, 256, 0, stream>>>(Wv, wbf + 2 * N_W, N_W);

  // 3 mats * 512 row-tiles * 32 col-chunks = 49152 waves / 4 per block
  qkv_proj<<<12288, 128, 0, stream>>>(xbf, wbf, bq, bk, bv, qbf, kbf, vtbf);
  // B*H*(S/64) blocks, 4 waves each sharing TDM-staged K/V tiles
  attn_fwd<<<1024, 128, 0, stream>>>(qbf, kbf, vtbf, out);
}
